// SelfAttention_54331336294755
// MI455X (gfx1250) — compile-verified, matrix operand order checked by reading, not run on hardware
//
#include <hip/hip_runtime.h>
#include <hip/hip_bf16.h>

// Problem constants (match reference).
constexpr int N_TOK   = 8192;
constexpr int DIM_K   = 512;  // input feature dim
constexpr int D_MODEL = 512;  // projected dim

typedef __attribute__((ext_vector_type(16))) __bf16 v16bf;
typedef __attribute__((ext_vector_type(8)))  __bf16 v8bf;
typedef __attribute__((ext_vector_type(4)))  __bf16 v4bf;
typedef __attribute__((ext_vector_type(8)))  float  v8f;
typedef __attribute__((ext_vector_type(4)))  float  v4f;
typedef __attribute__((ext_vector_type(4)))  unsigned int u32x4;
typedef __attribute__((ext_vector_type(8)))  int    i32x8;
typedef __attribute__((ext_vector_type(4)))  int    i32x4;

// ---------------------------------------------------------------------------
// BF16 WMMA helper: D = A(16x32) * B(32x16) + C, fp32 accumulate.
// ---------------------------------------------------------------------------
__device__ __forceinline__ v8f wmma_bf16(v16bf a, v16bf b, v8f c) {
  return __builtin_amdgcn_wmma_f32_16x16x32_bf16(
      /*neg_a=*/false, a, /*neg_b=*/false, b,
      /*c_mod=*/(short)0, c, /*reuse_a=*/false, /*reuse_b=*/false);
}

// Load one 16x32 bf16 fragment from an LDS tile stored row-major with row
// stride LDT (elements). Per ISA 7.12.2 (16-bit A layout):
//   lanes 0-15 : row=lane,    elements 0..7 = K0..7,  8..15 = K16..23
//   lanes 16-31: row=lane-16, elements 0..7 = K8..15, 8..15 = K24..31
template <int LDT>
__device__ __forceinline__ v16bf load_frag(const __bf16* base, int lane) {
  const __bf16* p = base + (lane & 15) * LDT + ((lane >> 4) << 3);
  v8bf lo = *(const v8bf*)p;
  v8bf hi = *(const v8bf*)(p + 16);
  return __builtin_shufflevector(lo, hi, 0, 1, 2, 3, 4, 5, 6, 7,
                                         8, 9, 10, 11, 12, 13, 14, 15);
}

// ---------------------------------------------------------------------------
// Tensor Data Mover: DMA one 32(contig) x 128(rows) bf16 tile from a
// [N_TOK x 512] bf16 tensor (row stride 512 elems) into LDS, padding each
// 64-byte tile row with 16 bytes so the LDS layout is 40-element rows
// (pad_interval code 3 = 16 DWORDs, pad_amount code 3 = 4 DWORDs).
// Tracked by TENSORcnt; issued by one wave only (TDM ignores EXEC).
// ---------------------------------------------------------------------------
__device__ __forceinline__ void tdm_load_tile_bf16(unsigned lds_addr,
                                                   const __bf16* gsrc) {
  unsigned long long ga = (unsigned long long)(size_t)gsrc;
  u32x4 g0;
  g0.x = 1u;                                     // count=1, user-mode D#
  g0.y = lds_addr;                               // lds_addr (bytes)
  g0.z = (unsigned)(ga & 0xFFFFFFFFull);         // global_addr[31:0]
  g0.w = (unsigned)((ga >> 32) & 0x1FFFFFFull)   // global_addr[56:32]
       | 0x80000000u;                            // type = 2 ("image")
  i32x8 g1;
  g1[0] = (1 << 16)                              // data_size = 2 bytes
        | (1 << 20)                              // pad_enable
        | (3 << 22)                              // pad_interval: 16 DWORDs
        | (3 << 25);                             // pad_amount:   4 DWORDs
  g1[1] = (DIM_K & 0xFFFF) << 16;                // tensor_dim0 lo16
  g1[2] = ((N_TOK & 0xFFFF) << 16) | ((DIM_K >> 16) & 0xFFFF);  // dim1 lo | dim0 hi
  g1[3] = (32 << 16) | ((N_TOK >> 16) & 0xFFFF); // tile_dim0=32 | dim1 hi16
  g1[4] = 128;                                   // tile_dim1 = 128 rows
  g1[5] = DIM_K;                                 // tensor_dim0_stride lo32
  g1[6] = 0;
  g1[7] = 0;
  i32x4 z4 = {0, 0, 0, 0};
#if __clang_major__ >= 23
  i32x8 z8 = {0, 0, 0, 0, 0, 0, 0, 0};
  __builtin_amdgcn_tensor_load_to_lds(g0, g1, z4, z4, z8, 0);
#else
  __builtin_amdgcn_tensor_load_to_lds(g0, g1, z4, z4, 0);
#endif
}

// ---------------------------------------------------------------------------
// Kernel 1: projection GEMM   out(bf16)[N,512] = A(f32)[N,512] @ W(f32)[512,512] + bias
// Tile 128x128, K-chunk 32, 8 waves (4x2), each wave 32x64 (2x4 WMMA tiles).
// ---------------------------------------------------------------------------
__global__ __launch_bounds__(256)
void proj_kernel(const float* __restrict__ A, const float* __restrict__ W,
                 const float* __restrict__ bias, __bf16* __restrict__ out) {
  constexpr int LD = 40;  // padded LDS row stride (keeps 16B alignment)
  __shared__ __bf16 sA[128 * LD];
  __shared__ __bf16 sB[128 * LD];
  const int tid = threadIdx.x, lane = tid & 31, wid = tid >> 5;
  const int wm = wid >> 1, wn = wid & 1;      // 4x2 wave grid
  const int n0 = blockIdx.x * 128;            // output column tile
  const int m0 = blockIdx.y * 128;            // output row tile

  v8f acc[2][4] = {};
  for (int kk0 = 0; kk0 < DIM_K; kk0 += 32) {
    // Stage A tile (128 rows x 32 k), fp32 -> bf16, row-major.
    for (int p = 0; p < 4; ++p) {
      int e = (p * 256 + tid) * 4;
      int r = e >> 5, c = e & 31;
      v4f x = *(const v4f*)(A + (size_t)(m0 + r) * DIM_K + kk0 + c);
      v4bf y = {(__bf16)x.x, (__bf16)x.y, (__bf16)x.z, (__bf16)x.w};
      *(v4bf*)(sA + r * LD + c) = y;
    }
    // Stage W tile (32 k x 128 n) TRANSPOSED into sB[n][k] so each lane can
    // read "row n" = column n of W (B-fragment layout).
    for (int p = 0; p < 4; ++p) {
      int e = (p * 256 + tid) * 4;
      int kk = e >> 7, n = e & 127;
      v4f x = *(const v4f*)(W + (size_t)(kk0 + kk) * D_MODEL + n0 + n);
      sB[(n + 0) * LD + kk] = (__bf16)x.x;
      sB[(n + 1) * LD + kk] = (__bf16)x.y;
      sB[(n + 2) * LD + kk] = (__bf16)x.z;
      sB[(n + 3) * LD + kk] = (__bf16)x.w;
    }
    __syncthreads();
    v16bf a[2], b[4];
    for (int i = 0; i < 2; ++i) a[i] = load_frag<LD>(sA + (wm * 32 + i * 16) * LD, lane);
    for (int j = 0; j < 4; ++j) b[j] = load_frag<LD>(sB + (wn * 64 + j * 16) * LD, lane);
    for (int i = 0; i < 2; ++i)
      for (int j = 0; j < 4; ++j)
        acc[i][j] = wmma_bf16(a[i], b[j], acc[i][j]);
    __syncthreads();
  }
  // Epilogue: add bias, convert to bf16. C layout: lane = col (mod 16),
  // vgpr r = row r (lanes 0-15) or row r+8 (lanes 16-31).
  const int half = lane >> 4, nl = lane & 15;
  for (int i = 0; i < 2; ++i)
    for (int j = 0; j < 4; ++j) {
      int n = n0 + wn * 64 + j * 16 + nl;
      float bv = bias[n];
      for (int r = 0; r < 8; ++r) {
        int row = m0 + wm * 32 + i * 16 + half * 8 + r;
        out[(size_t)row * D_MODEL + n] = (__bf16)(acc[i][j][r] + bv);
      }
    }
}

// ---------------------------------------------------------------------------
// Kernel 2: scores  S[m,n] = label[m,n] * scale * (q[m,:] . k[n,:])
// q@k^T: both fragments load row-major rows (row-per-lane) => B = k^T free.
// Tiles are DMA'd by the Tensor Data Mover (wave 0), double-buffered so the
// next chunk streams in while all 8 waves run WMMAs on the current one.
// ---------------------------------------------------------------------------
__global__ __launch_bounds__(256)
void scores_kernel(const __bf16* __restrict__ qb, const __bf16* __restrict__ kb,
                   const float* __restrict__ label, float* __restrict__ S) {
  constexpr int LD = 40;
  __shared__ __bf16 sQ[2][128 * LD];
  __shared__ __bf16 sK[2][128 * LD];
  const int tid = threadIdx.x, lane = tid & 31, wid = tid >> 5;
  const int wm = wid >> 1, wn = wid & 1;
  const int n0 = blockIdx.x * 128, m0 = blockIdx.y * 128;

  if (wid == 0) {  // preload chunk 0
    tdm_load_tile_bf16((unsigned)(size_t)&sQ[0][0], qb + (size_t)m0 * D_MODEL);
    tdm_load_tile_bf16((unsigned)(size_t)&sK[0][0], kb + (size_t)n0 * D_MODEL);
  }

  v8f acc[2][4] = {};
  int cur = 0;
  for (int kk0 = 0; kk0 < D_MODEL; kk0 += 32, cur ^= 1) {
    if (wid == 0) __builtin_amdgcn_s_wait_tensorcnt(0);
    __syncthreads();  // buf[cur] ready; all waves done reading buf[cur^1]
    if (wid == 0 && kk0 + 32 < D_MODEL) {
      tdm_load_tile_bf16((unsigned)(size_t)&sQ[cur ^ 1][0],
                         qb + (size_t)m0 * D_MODEL + kk0 + 32);
      tdm_load_tile_bf16((unsigned)(size_t)&sK[cur ^ 1][0],
                         kb + (size_t)n0 * D_MODEL + kk0 + 32);
    }
    v16bf a[2], b[4];
    for (int i = 0; i < 2; ++i)
      a[i] = load_frag<LD>(&sQ[cur][0] + (wm * 32 + i * 16) * LD, lane);
    for (int j = 0; j < 4; ++j)
      b[j] = load_frag<LD>(&sK[cur][0] + (wn * 64 + j * 16) * LD, lane);
    for (int i = 0; i < 2; ++i)
      for (int j = 0; j < 4; ++j)
        acc[i][j] = wmma_bf16(a[i], b[j], acc[i][j]);
    // No trailing barrier: fragment ds_loads are consumed by the WMMAs above,
    // and the next overwrite of buf[cur^1] is issued only after the top
    // barrier of the next iteration.
  }

  const float scale = 0.044194173824159216f;  // 1/sqrt(512)
  const int half = lane >> 4, nl = lane & 15;
  for (int i = 0; i < 2; ++i)
    for (int j = 0; j < 4; ++j) {
      int col = n0 + wn * 64 + j * 16 + nl;
      for (int r = 0; r < 8; ++r) {
        int row = m0 + wm * 32 + i * 16 + half * 8 + r;
        size_t idx = (size_t)row * N_TOK + col;
        S[idx] = acc[i][j][r] * scale * label[idx];
      }
    }
}

// ---------------------------------------------------------------------------
// Kernel 3: per-row max and sum(exp(x-max)). One block per row.
// ---------------------------------------------------------------------------
__global__ __launch_bounds__(256)
void rowstats_kernel(const float* __restrict__ S, float* __restrict__ rmax,
                     float* __restrict__ rsum) {
  __shared__ float red[256];
  const int row = blockIdx.x, tid = threadIdx.x;
  const float* rp = S + (size_t)row * N_TOK;
  float m = -3.402823466e38f;
  for (int c = tid * 4; c < N_TOK; c += 1024) {
    v4f x = *(const v4f*)(rp + c);
    m = fmaxf(m, fmaxf(fmaxf(x.x, x.y), fmaxf(x.z, x.w)));
  }
  red[tid] = m;
  __syncthreads();
  for (int s = 128; s > 0; s >>= 1) {
    if (tid < s) red[tid] = fmaxf(red[tid], red[tid + s]);
    __syncthreads();
  }
  const float rowm = red[0];
  __syncthreads();
  float sum = 0.f;
  for (int c = tid * 4; c < N_TOK; c += 1024) {
    v4f x = *(const v4f*)(rp + c);
    sum += __expf(x.x - rowm) + __expf(x.y - rowm) +
           __expf(x.z - rowm) + __expf(x.w - rowm);
  }
  red[tid] = sum;
  __syncthreads();
  for (int s = 128; s > 0; s >>= 1) {
    if (tid < s) red[tid] += red[tid + s];
    __syncthreads();
  }
  if (tid == 0) { rmax[row] = rowm; rsum[row] = red[0]; }
}

// ---------------------------------------------------------------------------
// Kernel 4: in-place softmax normalize: attn = exp(S - max[row]) / sum[row]
// ---------------------------------------------------------------------------
__global__ __launch_bounds__(256)
void normalize_kernel(float* __restrict__ S, const float* __restrict__ rmax,
                      const float* __restrict__ rsum) {
  size_t gid = (size_t)blockIdx.x * 256 + threadIdx.x;  // float4 index
  int row = (int)(gid >> 11);                           // 2048 float4 / row
  float m = rmax[row];
  float inv = 1.0f / rsum[row];
  float* p = S + gid * 4;
  v4f x = *(const v4f*)p;
  v4f y = {__expf(x.x - m) * inv, __expf(x.y - m) * inv,
           __expf(x.z - m) * inv, __expf(x.w - m) * inv};
  *(v4f*)p = y;
}

// ---------------------------------------------------------------------------
// Kernel 5: context = attn(f32->bf16) @ v(bf16).  BM=32, BN=512 so each block
// streams its 32 attn rows exactly once (268 MB total HBM); vb stays in L2.
// 8 waves (2x4): each wave 16 rows x 128 cols = 8 WMMA tiles.
// v tile is transposed through LDS so lane n reads column n of v.
// ---------------------------------------------------------------------------
__global__ __launch_bounds__(256)
void context_kernel(const float* __restrict__ attn, const __bf16* __restrict__ vb,
                    float* __restrict__ out) {
  constexpr int LD = 40;
  __shared__ __bf16 sA[32 * LD];    // 32 rows x 32 k
  __shared__ __bf16 sB[512 * LD];   // 512 n x 32 k  (v transposed)
  const int tid = threadIdx.x, lane = tid & 31, wid = tid >> 5;
  const int wm = wid >> 2, wn = wid & 3;  // 2x4 wave grid
  const int m0 = blockIdx.x * 32;

  v8f acc[8] = {};
  for (int kk0 = 0; kk0 < N_TOK; kk0 += 32) {
    // A: attn[m0..+32][kk0..+32] fp32 -> bf16 (one float4 per thread).
    {
      int e = tid * 4;
      int r = e >> 5, c = e & 31;
      const float* ap = attn + (size_t)(m0 + r) * N_TOK + kk0 + c;
      v4f x = *(const v4f*)ap;
      if (kk0 + 32 < N_TOK) __builtin_prefetch(ap + 32, 0, 1);  // global_prefetch_b8
      v4bf y = {(__bf16)x.x, (__bf16)x.y, (__bf16)x.z, (__bf16)x.w};
      *(v4bf*)(sA + r * LD + c) = y;
    }
    // B: vb[kk0..+32][0..512] transposed into sB[n][k].
    for (int p = 0; p < 8; ++p) {
      int e = (p * 256 + tid) * 8;
      int kk = e >> 9, n = e & 511;
      v8bf x = *(const v8bf*)(vb + (size_t)(kk0 + kk) * D_MODEL + n);
      for (int j = 0; j < 8; ++j) sB[(n + j) * LD + kk] = x[j];
    }
    __syncthreads();
    v16bf a = load_frag<LD>(sA + (wm * 16) * LD, lane);
    for (int j = 0; j < 8; ++j) {
      v16bf b = load_frag<LD>(sB + (wn * 128 + j * 16) * LD, lane);
      acc[j] = wmma_bf16(a, b, acc[j]);
    }
    __syncthreads();
  }
  const int half = lane >> 4, nl = lane & 15;
  for (int j = 0; j < 8; ++j) {
    int n = wn * 128 + j * 16 + nl;
    for (int r = 0; r < 8; ++r) {
      int row = m0 + wm * 16 + half * 8 + r;
      out[(size_t)row * D_MODEL + n] = acc[j][r];
    }
  }
}

// ---------------------------------------------------------------------------
// Host-side launcher.
// Inputs: 0=Q 1=K 2=V 3=label_graph 4=Wq 5=bq 6=Wv 7=bv (all fp32).
// d_out: context [8192*512] fp32, then attn [8192*8192] fp32.
// ws: qb/kb/vb bf16 (3 x 8 MB) + row_max/row_sum (2 x 32 KB).
// ---------------------------------------------------------------------------
extern "C" void kernel_launch(void* const* d_in, const int* in_sizes, int n_in,
                              void* d_out, int out_size, void* d_ws, size_t ws_size,
                              hipStream_t stream) {
  const float* Q     = (const float*)d_in[0];
  const float* K     = (const float*)d_in[1];
  const float* V     = (const float*)d_in[2];
  const float* label = (const float*)d_in[3];
  const float* Wq    = (const float*)d_in[4];
  const float* bq    = (const float*)d_in[5];
  const float* Wv    = (const float*)d_in[6];
  const float* bv    = (const float*)d_in[7];

  __bf16* qb = (__bf16*)d_ws;
  __bf16* kb = qb + (size_t)N_TOK * D_MODEL;
  __bf16* vb = kb + (size_t)N_TOK * D_MODEL;
  float* rmax = (float*)(vb + (size_t)N_TOK * D_MODEL);
  float* rsum = rmax + N_TOK;

  float* ctx  = (float*)d_out;
  float* attn = ctx + (size_t)N_TOK * D_MODEL;

  dim3 blk(256);
  // Projections (K uses Wq/bq per the reference's quirk).
  proj_kernel<<<dim3(D_MODEL / 128, N_TOK / 128), blk, 0, stream>>>(Q, Wq, bq, qb);
  proj_kernel<<<dim3(D_MODEL / 128, N_TOK / 128), blk, 0, stream>>>(K, Wq, bq, kb);
  proj_kernel<<<dim3(D_MODEL / 128, N_TOK / 128), blk, 0, stream>>>(V, Wv, bv, vb);
  // Masked scores -> attn region of d_out (pre-softmax).
  scores_kernel<<<dim3(N_TOK / 128, N_TOK / 128), blk, 0, stream>>>(qb, kb, label, attn);
  // Safe softmax.
  rowstats_kernel<<<dim3(N_TOK), blk, 0, stream>>>(attn, rmax, rsum);
  normalize_kernel<<<dim3((size_t)N_TOK * N_TOK / 4 / 256), blk, 0, stream>>>(attn, rmax, rsum);
  // context = attn @ v
  context_kernel<<<dim3(N_TOK / 32), blk, 0, stream>>>(attn, vb, ctx);
}
